// Pointnet2Backbone_13984413515828
// MI455X (gfx1250) — matrix operand order repaired and checked by reading, not compiled, not run on hardware
//
#include <hip/hip_runtime.h>

typedef __attribute__((ext_vector_type(16))) _Float16 v16h;
typedef __attribute__((ext_vector_type(8)))  float    v8f;

// ---------------------------------------------------------------------
// gfx1250 async global->LDS path (guarded; falls back to sync staging)
// Builtin signature (from probe diagnostics):
//   void __builtin_amdgcn_global_load_async_to_lds_b128(
//        int __vector_size__(16) AS1* src, int __vector_size__(16) AS3* dst,
//        imm int offset, imm int cpol)
// ---------------------------------------------------------------------
#if defined(__gfx1250__) && __has_builtin(__builtin_amdgcn_global_load_async_to_lds_b128)
#define HAVE_ASYNC_LDS 1
#else
#define HAVE_ASYNC_LDS 0
#endif

#if HAVE_ASYNC_LDS
typedef int gvec4i __attribute__((vector_size(16)));   // GCC-style vector (matches "V4i")
typedef __attribute__((address_space(1))) gvec4i as1_v4i;
typedef __attribute__((address_space(3))) gvec4i as3_v4i;
__device__ __forceinline__ as1_v4i* to_as1(const void* p) {
  return (as1_v4i*)reinterpret_cast<unsigned long long>(p);
}
__device__ __forceinline__ as3_v4i* to_as3(const void* p) {
  // generic LDS pointer: low 32 bits are the LDS offset
  return (as3_v4i*)(unsigned int)reinterpret_cast<unsigned long long>(p);
}
#if __has_builtin(__builtin_amdgcn_s_wait_asynccnt)
#define WAIT_ASYNCCNT(n) __builtin_amdgcn_s_wait_asynccnt(n)
#else
#define WAIT_ASYNCCNT(n) asm volatile("s_wait_asynccnt %0" :: "i"(n) : "memory")
#endif
#endif

// =====================================================================
// WMMA GEMM:  Y[M x O](f16) = relu( (X[M x K](f16) . W[O x K]^T(f16)) * g + b )
// M % 64 == 0, O % 128 == 0, K % 32 == 0.  ldX = ldW = K, ldY = O.
// Block: 256 threads = 8 waves, 64x128 output tile, wave grid 4(M) x 2(N),
// each wave holds four 16x16 f32 accumulators -> 4 v_wmma per K step,
// one shared A fragment per wave. Async double-buffered LDS staging.
// =====================================================================
__global__ __launch_bounds__(256) void gemm_cbr_wmma(
    const _Float16* __restrict__ X, const _Float16* __restrict__ W,
    const float* __restrict__ gain, const float* __restrict__ bias,
    _Float16* __restrict__ Y, int M, int K, int O)
{
  __shared__ __align__(32) _Float16 ldsA[2][4 * 512];   // 64x32 A tile / buffer
  __shared__ __align__(32) _Float16 ldsB[2][8 * 512];   // 32x128 B tile / buffer

  const int tid  = threadIdx.x;
  const int lane = tid & 31;
  const int wave = tid >> 5;
  const int mw   = wave >> 1;        // 0..3 : 16-row block
  const int nw   = wave & 1;         // 0..1 : 64-col half
  const int bm   = blockIdx.y * 64;
  const int bn   = blockIdx.x * 128;

  // --- A staging: thread -> row (tid>>2, 0..63), 8 halfs at segA ---
  const int rowA = tid >> 2;
  const int segA = (tid & 3) * 8;
  // A fragment layout (16x32): lane = m + 16*(k>>4), slot = k&15: contiguous 16B per thread
  const int dstA = (rowA >> 4) * 512
                 + ((rowA & 15) + ((segA >> 4) << 4)) * 16
                 + (((segA >> 3) & 1) << 3);
  const _Float16* gA = X + (size_t)(bm + rowA) * K + segA;

  // --- B staging: thread -> row (tid>>1, 0..127), 16 halfs at segB ---
  const int rowB = tid >> 1;
  const int segB = (tid & 1) * 16;
  // B fragment layout (32x16): lane = n + 16*(k>>4), slot = k&15 -> 32B per thread
  const int dstB = (rowB >> 4) * 512
                 + ((rowB & 15) + ((segB >> 4) << 4)) * 16;
  const _Float16* gB = W + (size_t)(bn + rowB) * K + segB;

  const int T = K >> 5;   // number of 32-wide K tiles (>= 1)

  v8f acc[4] = {};

#if HAVE_ASYNC_LDS
  // prologue: issue tiles 0 and 1
  __builtin_amdgcn_global_load_async_to_lds_b128(to_as1(gA), to_as3(&ldsA[0][dstA]), 0, 0);
  __builtin_amdgcn_global_load_async_to_lds_b128(to_as1(gB), to_as3(&ldsB[0][dstB]), 0, 0);
  __builtin_amdgcn_global_load_async_to_lds_b128(to_as1(gB + 8), to_as3(&ldsB[0][dstB + 8]), 0, 0);
  if (T > 1) {
    __builtin_amdgcn_global_load_async_to_lds_b128(to_as1(gA + 32), to_as3(&ldsA[1][dstA]), 0, 0);
    __builtin_amdgcn_global_load_async_to_lds_b128(to_as1(gB + 32), to_as3(&ldsB[1][dstB]), 0, 0);
    __builtin_amdgcn_global_load_async_to_lds_b128(to_as1(gB + 40), to_as3(&ldsB[1][dstB + 8]), 0, 0);
  }
  for (int t = 0; t < T; ++t) {
    const int buf = t & 1;
    // async loads complete in order: <=3 pending means tile t has landed
    if (t + 1 < T) { WAIT_ASYNCCNT(3); } else { WAIT_ASYNCCNT(0); }
    __syncthreads();
    const v16h a = *(const v16h*)&ldsA[buf][mw * 512 + lane * 16];
    const _Float16* lB = &ldsB[buf][nw * 2048 + lane * 16];
#pragma unroll
    for (int f = 0; f < 4; ++f) {
      const v16h b = *(const v16h*)(lB + f * 512);
      acc[f] = __builtin_amdgcn_wmma_f32_16x16x32_f16(
          false, a, false, b, (short)0, acc[f], false, false);
    }
    __syncthreads();   // all waves done reading buf before it is overwritten
    if (t + 2 < T) {
      const _Float16* a2 = gA + ((t + 2) << 5);
      const _Float16* b2 = gB + ((t + 2) << 5);
      __builtin_amdgcn_global_load_async_to_lds_b128(to_as1(a2), to_as3(&ldsA[buf][dstA]), 0, 0);
      __builtin_amdgcn_global_load_async_to_lds_b128(to_as1(b2), to_as3(&ldsB[buf][dstB]), 0, 0);
      __builtin_amdgcn_global_load_async_to_lds_b128(to_as1(b2 + 8), to_as3(&ldsB[buf][dstB + 8]), 0, 0);
    }
  }
#else
  for (int k0 = 0; k0 < K; k0 += 32) {
    const uint4 ta  = *(const uint4*)(gA + k0);
    const uint4 tb0 = *(const uint4*)(gB + k0);
    const uint4 tb1 = *(const uint4*)(gB + k0 + 8);
    if (k0 + 32 < K) {
      __builtin_prefetch((const void*)(gA + k0 + 32), 0, 3);
      __builtin_prefetch((const void*)(gB + k0 + 32), 0, 3);
    }
    *(uint4*)&ldsA[0][dstA]     = ta;
    *(uint4*)&ldsB[0][dstB]     = tb0;
    *(uint4*)&ldsB[0][dstB + 8] = tb1;
    __syncthreads();
    const v16h a = *(const v16h*)&ldsA[0][mw * 512 + lane * 16];
    const _Float16* lB = &ldsB[0][nw * 2048 + lane * 16];
#pragma unroll
    for (int f = 0; f < 4; ++f) {
      const v16h b = *(const v16h*)(lB + f * 512);
      acc[f] = __builtin_amdgcn_wmma_f32_16x16x32_f16(
          false, a, false, b, (short)0, acc[f], false, false);
    }
    __syncthreads();
  }
#endif

  // C/D layout: lanes 0-15 -> M=r, lanes 16-31 -> M=8+r ; N = lane&15
  const int lhi  = lane >> 4;
  const int nloc = lane & 15;
#pragma unroll
  for (int f = 0; f < 4; ++f) {
    const int n = bn + nw * 64 + f * 16 + nloc;
    const float gv = gain[n], bv = bias[n];
#pragma unroll
    for (int r = 0; r < 8; ++r) {
      const int m = bm + mw * 16 + lhi * 8 + r;
      float y = acc[f][r] * gv + bv;
      y = y > 0.f ? y : 0.f;
      Y[(size_t)m * O + n] = (_Float16)y;
    }
  }
}

// =====================================================================
// Farthest point sampling: one 1024-thread block, block-wide argmax rounds
// =====================================================================
__global__ __launch_bounds__(1024) void fps_kernel(
    const float* __restrict__ xyz, int N, int npoint, int* __restrict__ inds)
{
  __shared__ float sval[1024];
  __shared__ int   sidx[1024];
  __shared__ int   scur;
  const int tid = threadIdx.x;
  float px[16], py[16], pz[16], dist[16];
#pragma unroll
  for (int i = 0; i < 16; ++i) {
    const int n = tid + (i << 10);
    if (n < N) {
      px[i] = xyz[n * 3 + 0]; py[i] = xyz[n * 3 + 1]; pz[i] = xyz[n * 3 + 2];
      dist[i] = 1e10f;
    } else { px[i] = py[i] = pz[i] = 0.f; dist[i] = -1.f; }
  }
  if (tid == 0) scur = 0;
  __syncthreads();
  for (int t = 0; t < npoint; ++t) {
    const int cur = scur;
    if (tid == 0) inds[t] = cur;
    const float cx = xyz[cur * 3 + 0], cy = xyz[cur * 3 + 1], cz = xyz[cur * 3 + 2];
    float best = -2.f; int bestI = 0;
#pragma unroll
    for (int i = 0; i < 16; ++i) {
      const int n = tid + (i << 10);
      if (n < N) {
        const float dx = px[i] - cx, dy = py[i] - cy, dz = pz[i] - cz;
        const float d = dx * dx + dy * dy + dz * dz;
        if (d < dist[i]) dist[i] = d;
        if (dist[i] > best) { best = dist[i]; bestI = n; }
      }
    }
    sval[tid] = best; sidx[tid] = bestI;
    __syncthreads();
    for (int s = 512; s > 0; s >>= 1) {
      if (tid < s) {
        const float ov = sval[tid + s]; const int oi = sidx[tid + s];
        if (ov > sval[tid] || (ov == sval[tid] && oi < sidx[tid])) {
          sval[tid] = ov; sidx[tid] = oi;
        }
      }
      __syncthreads();
    }
    if (tid == 0) scur = sidx[0];
    __syncthreads();
  }
}

__global__ void gather_xyz_kernel(const float* __restrict__ xyz,
                                  const int* __restrict__ inds, int S,
                                  float* __restrict__ out)
{
  const int t = blockIdx.x * blockDim.x + threadIdx.x;
  if (t >= S) return;
  const int n = inds[t];
  out[t * 3 + 0] = xyz[n * 3 + 0];
  out[t * 3 + 1] = xyz[n * 3 + 1];
  out[t * 3 + 2] = xyz[n * 3 + 2];
}

// First nsample indices (in index order) within radius; pad with first (or 0)
__global__ void ball_query_kernel(float r2, int ns, const float* __restrict__ xyz,
                                  int Nin, const float* __restrict__ q, int S,
                                  int* __restrict__ idx)
{
  const int s = blockIdx.x * blockDim.x + threadIdx.x;
  if (s >= S) return;
  const float qx = q[s * 3 + 0], qy = q[s * 3 + 1], qz = q[s * 3 + 2];
  int* out = idx + (size_t)s * ns;
  int cnt = 0;
  for (int n = 0; n < Nin; ++n) {
    const float dx = xyz[n * 3 + 0] - qx, dy = xyz[n * 3 + 1] - qy, dz = xyz[n * 3 + 2] - qz;
    if (dx * dx + dy * dy + dz * dz < r2) {
      out[cnt++] = n;
      if (cnt == ns) break;
    }
  }
  const int first = (cnt > 0) ? out[0] : 0;
  for (int j = cnt; j < ns; ++j) out[j] = first;
}

// g = concat((xyz[idx]-new_xyz)/radius, feats[idx]) -> f16, K padded with zeros
__global__ void build_group_kernel(const float* __restrict__ xyz,
                                   const float* __restrict__ new_xyz,
                                   const _Float16* __restrict__ feats, int C,
                                   const int* __restrict__ idx, int nsample,
                                   float inv_r, int Kpad, int rows,
                                   _Float16* __restrict__ G)
{
  const int row = blockIdx.x * blockDim.x + threadIdx.x;
  if (row >= rows) return;
  const int s = row / nsample;
  const int n = idx[row];
  _Float16* dst = G + (size_t)row * Kpad;
  dst[0] = (_Float16)((xyz[n * 3 + 0] - new_xyz[s * 3 + 0]) * inv_r);
  dst[1] = (_Float16)((xyz[n * 3 + 1] - new_xyz[s * 3 + 1]) * inv_r);
  dst[2] = (_Float16)((xyz[n * 3 + 2] - new_xyz[s * 3 + 2]) * inv_r);
  const _Float16* src = feats + (size_t)n * C;
  for (int c = 0; c < C; ++c) dst[3 + c] = src[c];
  for (int c = 3 + C; c < Kpad; ++c) dst[c] = (_Float16)0.0f;
}

__global__ void maxpool_kernel(const _Float16* __restrict__ Y, int S, int ns,
                               _Float16* __restrict__ out)
{
  const int t = blockIdx.x * blockDim.x + threadIdx.x;
  if (t >= S * 256) return;
  const int s = t >> 8, c = t & 255;
  float m = -1e30f;
  const _Float16* base = Y + (size_t)s * ns * 256 + c;
  for (int j = 0; j < ns; ++j) m = fmaxf(m, (float)base[(size_t)j * 256]);
  out[t] = (_Float16)m;
}

// 3-NN + inverse-distance weights
__global__ void knn3_kernel(const float* __restrict__ xa, int na,
                            const float* __restrict__ xb, int nb,
                            int* __restrict__ kidx, float* __restrict__ kw)
{
  const int i = blockIdx.x * blockDim.x + threadIdx.x;
  if (i >= na) return;
  const float qx = xa[i * 3 + 0], qy = xa[i * 3 + 1], qz = xa[i * 3 + 2];
  float d0 = 3e38f, d1 = 3e38f, dd2 = 3e38f;
  int i0 = 0, i1 = 0, i2 = 0;
  for (int n = 0; n < nb; ++n) {
    const float dx = xb[n * 3 + 0] - qx, dy = xb[n * 3 + 1] - qy, dz = xb[n * 3 + 2] - qz;
    const float d = dx * dx + dy * dy + dz * dz;
    if (d < d0)       { dd2 = d1; i2 = i1; d1 = d0; i1 = i0; d0 = d; i0 = n; }
    else if (d < d1)  { dd2 = d1; i2 = i1; d1 = d; i1 = n; }
    else if (d < dd2) { dd2 = d; i2 = n; }
  }
  const float w0 = 1.f / (d0 + 1e-8f), w1 = 1.f / (d1 + 1e-8f), w2 = 1.f / (dd2 + 1e-8f);
  const float inv = 1.f / (w0 + w1 + w2);
  kidx[i * 3 + 0] = i0; kidx[i * 3 + 1] = i1; kidx[i * 3 + 2] = i2;
  kw[i * 3 + 0] = w0 * inv; kw[i * 3 + 1] = w1 * inv; kw[i * 3 + 2] = w2 * inv;
}

// Xo[i] = concat(featsA[i] (Ca ch), sum_k w_k * featsB[idx_k] (256 ch))
__global__ void concat_interp_kernel(const _Float16* __restrict__ featsA, int Ca,
                                     const _Float16* __restrict__ featsB,
                                     const int* __restrict__ kidx,
                                     const float* __restrict__ kw, int n1,
                                     _Float16* __restrict__ Xo)
{
  const int Kc = Ca + 256;
  const size_t t = (size_t)blockIdx.x * blockDim.x + threadIdx.x;
  if (t >= (size_t)n1 * Kc) return;
  const int i = (int)(t / Kc), c = (int)(t % Kc);
  _Float16 v;
  if (c < Ca) v = featsA[(size_t)i * Ca + c];
  else {
    const int cc = c - Ca;
    const int i0 = kidx[i * 3 + 0], i1 = kidx[i * 3 + 1], i2 = kidx[i * 3 + 2];
    const float val = kw[i * 3 + 0] * (float)featsB[(size_t)i0 * 256 + cc]
                    + kw[i * 3 + 1] * (float)featsB[(size_t)i1 * 256 + cc]
                    + kw[i * 3 + 2] * (float)featsB[(size_t)i2 * 256 + cc];
    v = (_Float16)val;
  }
  Xo[t] = v;
}

__global__ void concat2_kernel(const _Float16* __restrict__ A,
                               const _Float16* __restrict__ Bf, int n,
                               _Float16* __restrict__ Xo)
{
  const size_t t = (size_t)blockIdx.x * blockDim.x + threadIdx.x;
  if (t >= (size_t)n * 512) return;
  const int i = (int)(t >> 9), c = (int)(t & 511);
  Xo[t] = (c < 256) ? A[(size_t)i * 256 + c] : Bf[(size_t)i * 256 + (c - 256)];
}

// 6-class linear head + log_softmax, f32 output
__global__ void head_kernel(const _Float16* __restrict__ X,
                            const float* __restrict__ W,
                            const float* __restrict__ b,
                            float* __restrict__ out, int Npts)
{
  const int i = blockIdx.x * blockDim.x + threadIdx.x;
  if (i >= Npts) return;
  float acc[6];
#pragma unroll
  for (int o = 0; o < 6; ++o) acc[o] = b[o];
  const _Float16* xr = X + (size_t)i * 256;
  for (int c = 0; c < 256; ++c) {
    const float x = (float)xr[c];
#pragma unroll
    for (int o = 0; o < 6; ++o) acc[o] += x * W[o * 256 + c];
  }
  float m = acc[0];
#pragma unroll
  for (int o = 1; o < 6; ++o) m = fmaxf(m, acc[o]);
  float s = 0.f;
#pragma unroll
  for (int o = 0; o < 6; ++o) s += __expf(acc[o] - m);
  const float lse = __logf(s);
#pragma unroll
  for (int o = 0; o < 6; ++o) out[(size_t)i * 6 + o] = acc[o] - m - lse;
}

__global__ void convert_w_kernel(const float* __restrict__ W,
                                 _Float16* __restrict__ Wh, int O, int K, int Kpad)
{
  const int t = blockIdx.x * blockDim.x + threadIdx.x;
  if (t >= O * Kpad) return;
  const int o = t / Kpad, k = t % Kpad;
  Wh[t] = (k < K) ? (_Float16)W[(size_t)o * K + k] : (_Float16)0.0f;
}

__global__ void extract_kernel(const float* __restrict__ pc,
                               float* __restrict__ xyz,
                               _Float16* __restrict__ f0h, int N)
{
  const int n = blockIdx.x * blockDim.x + threadIdx.x;
  if (n >= N) return;
  const float* row = pc + (size_t)n * 163;
  xyz[n * 3 + 0] = row[0]; xyz[n * 3 + 1] = row[1]; xyz[n * 3 + 2] = row[2];
  _Float16* d = f0h + (size_t)n * 160;
  for (int c = 0; c < 160; ++c) d[c] = (_Float16)row[3 + c];
}

// =====================================================================
// Host orchestration
// =====================================================================
// d_in flattening (JAX pytree: dict keys sorted alphabetically):
// 0: pointcloud (16384 x 163)
// 1-6: fp2 (W,g,b)x2   7-12: fp3   13-18: fp4   19-21: gf
// 22: gp_W 23: gp_b    24-29: lf   30-32: lgp1  33: lgp_W 34: lgp_b
// 35: lp_W 36: lp_b    37-45: sa1  46-54: sa2   55-63: sa3
struct LayerDef { int wi, gi, bi, O, K, Kpad; };
static const LayerDef LDEF[19] = {
  {24,25,26, 256,160,160},  // 0  lf0
  {27,28,29, 256,256,256},  // 1  lf1
  {37,38,39, 256,163,192},  // 2  sa1_0
  {40,41,42, 256,256,256},  // 3  sa1_1
  {43,44,45, 256,256,256},  // 4  sa1_2
  {46,47,48, 256,259,288},  // 5  sa2_0
  {49,50,51, 256,256,256},  // 6  sa2_1
  {52,53,54, 256,256,256},  // 7  sa2_2
  {55,56,57, 256,259,288},  // 8  sa3_0
  {58,59,60, 256,256,256},  // 9  sa3_1
  {61,62,63, 256,256,256},  // 10 sa3_2
  {1,2,3,    256,512,512},  // 11 fp2_0
  {4,5,6,    256,256,256},  // 12 fp2_1
  {7,8,9,    256,512,512},  // 13 fp3_0
  {10,11,12, 256,256,256},  // 14 fp3_1
  {13,14,15, 256,416,416},  // 15 fp4_0
  {16,17,18, 256,256,256},  // 16 fp4_1
  {19,20,21, 256,256,256},  // 17 gf0
  {30,31,32, 256,512,512},  // 18 lgp1_0
};

extern "C" void kernel_launch(void* const* d_in, const int* in_sizes, int n_in,
                              void* d_out, int out_size, void* d_ws, size_t ws_size,
                              hipStream_t stream)
{
  (void)in_sizes; (void)n_in; (void)out_size; (void)ws_size;
  const int N = 16384;
  char* ws = (char*)d_ws;
  size_t off = 0;
  auto alloc = [&](size_t bytes) -> void* {
    off = (off + 255) & ~(size_t)255;
    void* p = (void*)(ws + off);
    off += bytes;
    return p;
  };

  const float* pc = (const float*)d_in[0];

  float*    xyz0 = (float*)alloc((size_t)N * 3 * 4);
  _Float16* f0h  = (_Float16*)alloc((size_t)N * 160 * 2);

  _Float16* Wh[19];
  for (int l = 0; l < 19; ++l) {
    const LayerDef& L = LDEF[l];
    Wh[l] = (_Float16*)alloc((size_t)L.O * L.Kpad * 2);
    const int tot = L.O * L.Kpad;
    convert_w_kernel<<<(tot + 255) / 256, 256, 0, stream>>>(
        (const float*)d_in[L.wi], Wh[l], L.O, L.K, L.Kpad);
  }

  _Float16* bufA = (_Float16*)alloc((size_t)262144 * 288 * 2);  // widest input buffer
  _Float16* bufB = (_Float16*)alloc((size_t)262144 * 256 * 2);

  float* xyz1 = (float*)alloc((size_t)8192 * 3 * 4);
  float* xyz2 = (float*)alloc((size_t)4096 * 3 * 4);
  float* xyz3 = (float*)alloc((size_t)2048 * 3 * 4);
  _Float16* f1 = (_Float16*)alloc((size_t)8192 * 256 * 2);
  _Float16* f2 = (_Float16*)alloc((size_t)4096 * 256 * 2);
  _Float16* f3 = (_Float16*)alloc((size_t)2048 * 256 * 2);
  _Float16* fA = (_Float16*)alloc((size_t)4096 * 256 * 2);
  _Float16* fB = (_Float16*)alloc((size_t)8192 * 256 * 2);
  _Float16* fC = (_Float16*)alloc((size_t)N * 256 * 2);
  _Float16* lfbuf = (_Float16*)alloc((size_t)N * 256 * 2);
  _Float16* gfbuf = (_Float16*)alloc((size_t)N * 256 * 2);
  int*   inds = (int*)alloc((size_t)8192 * 4);
  int*   idxb = (int*)alloc((size_t)262144 * 4);
  int*   kidx = (int*)alloc((size_t)N * 3 * 4);
  float* kw   = (float*)alloc((size_t)N * 3 * 4);

  auto gemm = [&](const _Float16* X, int l, _Float16* Y, int M) {
    const LayerDef& L = LDEF[l];
    dim3 grid((unsigned)(L.O / 128), (unsigned)(M / 64));
    gemm_cbr_wmma<<<grid, 256, 0, stream>>>(
        X, Wh[l], (const float*)d_in[L.gi], (const float*)d_in[L.bi],
        Y, M, L.Kpad, L.O);
  };

  // ---- extract xyz0 / f0 (f16) ----
  extract_kernel<<<(N + 255) / 256, 256, 0, stream>>>(pc, xyz0, f0h, N);

  // ---- lf MLP + local head ----
  gemm(f0h, 0, bufB, N);
  gemm(bufB, 1, lfbuf, N);
  head_kernel<<<(N + 255) / 256, 256, 0, stream>>>(
      lfbuf, (const float*)d_in[35], (const float*)d_in[36],
      (float*)d_out + (size_t)N * 6, N);

  // ---- set-abstraction stage ----
  auto sa = [&](const float* xin, int Nin, const _Float16* fin, int Cin,
                int S, float radius, int ns, int l0,
                float* xout, _Float16* fout) {
    fps_kernel<<<1, 1024, 0, stream>>>(xin, Nin, S, inds);
    gather_xyz_kernel<<<(S + 255) / 256, 256, 0, stream>>>(xin, inds, S, xout);
    ball_query_kernel<<<(S + 255) / 256, 256, 0, stream>>>(
        radius * radius, ns, xin, Nin, xout, S, idxb);
    const int rows = S * ns;
    const int Kpad = LDEF[l0].Kpad;
    build_group_kernel<<<(rows + 255) / 256, 256, 0, stream>>>(
        xin, xout, fin, Cin, idxb, ns, 1.0f / radius, Kpad, rows, bufA);
    gemm(bufA, l0,     bufB, rows);
    gemm(bufB, l0 + 1, bufA, rows);
    gemm(bufA, l0 + 2, bufB, rows);
    maxpool_kernel<<<(unsigned)(((size_t)S * 256 + 255) / 256), 256, 0, stream>>>(
        bufB, S, ns, fout);
  };

  sa(xyz0, N,    f0h, 160, 8192, 1.0f,  16, 2, xyz1, f1);
  sa(xyz1, 8192, f1,  256, 4096, 5.0f,  64, 5, xyz2, f2);
  sa(xyz2, 4096, f2,  256, 2048, 15.0f, 64, 8, xyz3, f3);

  // ---- feature propagation stage ----
  auto fp = [&](const float* xa, int na, const float* xb, int nb,
                const _Float16* fa, int Ca, const _Float16* fb,
                int l0, _Float16* fout) {
    knn3_kernel<<<(na + 255) / 256, 256, 0, stream>>>(xa, na, xb, nb, kidx, kw);
    const size_t tot = (size_t)na * (Ca + 256);
    concat_interp_kernel<<<(unsigned)((tot + 255) / 256), 256, 0, stream>>>(
        fa, Ca, fb, kidx, kw, na, bufA);
    gemm(bufA, l0,     bufB, na);
    gemm(bufB, l0 + 1, fout, na);
  };

  fp(xyz2, 4096, xyz3, 2048, f2,  256, f3, 11, fA);
  fp(xyz1, 8192, xyz2, 4096, f1,  256, fA, 13, fB);
  fp(xyz0, N,    xyz1, 8192, f0h, 160, fB, 15, fC);

  // ---- gf MLP + global head ----
  gemm(fC, 17, gfbuf, N);
  head_kernel<<<(N + 255) / 256, 256, 0, stream>>>(
      gfbuf, (const float*)d_in[22], (const float*)d_in[23],
      (float*)d_out + (size_t)2 * N * 6, N);

  // ---- combined head ----
  concat2_kernel<<<(unsigned)(((size_t)N * 512 + 255) / 256), 256, 0, stream>>>(
      lfbuf, gfbuf, N, bufA);
  gemm(bufA, 18, bufB, N);
  head_kernel<<<(N + 255) / 256, 256, 0, stream>>>(
      bufB, (const float*)d_in[33], (const float*)d_in[34], (float*)d_out, N);
}
